// TensorTrainAdapter_58574763982868
// MI455X (gfx1250) — matrix operand order repaired and checked by reading, not compiled, not run on hardware
//
#include <hip/hip_runtime.h>
#include <stdint.h>

// CDNA5 / gfx1250: wave32, WMMA bf16 16x16x32 with f32 accumulate.
typedef __attribute__((ext_vector_type(16))) __bf16 v16bf;
typedef __attribute__((ext_vector_type(16))) float  v16f;
typedef __attribute__((ext_vector_type(8)))  float  v8f;
typedef __attribute__((ext_vector_type(4)))  float  f4;

__device__ __forceinline__ unsigned short f32_to_bf16_rne(float f) {
    unsigned int u = __float_as_uint(f);
    u += 0x7FFFu + ((u >> 16) & 1u);   // round-to-nearest-even
    return (unsigned short)(u >> 16);
}

union AFrag {
    v16bf bf;
    uint4 q[2];
};

// ---------------------------------------------------------------------------
// Kernel 0: contract TT cores -> dense Wd (1024x64) and Wu (64x1024), quantize
// to bf16, and scatter into this file's B-fragment order (contiguous k):
// fragment (t = n-tile, s = k-step of 32): lane l = n' + 16*h holds 16 bf16
// values B[32s + 16h + q][16t + n'], q = 0..15  (same k order as the A fragments).
// ---------------------------------------------------------------------------
__global__ void tt_pack_kernel(const float* __restrict__ d1, const float* __restrict__ d2,
                               const float* __restrict__ d3, const float* __restrict__ u1,
                               const float* __restrict__ u2, const float* __restrict__ u3,
                               unsigned short* __restrict__ wd_pack,   // 65536 bf16 (128 KB)
                               unsigned short* __restrict__ wu_pack) { // 65536 bf16 (128 KB)
    int gid = blockIdx.x * blockDim.x + threadIdx.x;
    if (gid < 65536) {
        // Wd[i][j], i = (i1,i2,i3) over (8,16,8), j = (j1,j2,j3) over (4,4,4)
        int i = gid >> 6, j = gid & 63;
        int i1 = i >> 7, i2 = (i >> 3) & 15, i3 = i & 7;
        int j1 = j >> 4, j2 = (j >> 2) & 3,  j3 = j & 3;
        float acc = 0.f;
        for (int a = 0; a < 8; ++a) {
            float pa = d1[(i1 * 4 + j1) * 8 + a];
            const float* d2p = d2 + ((a * 16 + i2) * 4 + j2) * 8;
            for (int b = 0; b < 8; ++b)
                acc += pa * d2p[b] * d3[(b * 8 + i3) * 4 + j3];
        }
        int t = j >> 4, np = j & 15;
        int s = i >> 5, r = i & 31, h = r >> 4, q = r & 15;
        int lane = np + 16 * h;
        wd_pack[(((t * 32 + s) * 32 + lane) << 4) + q] = f32_to_bf16_rne(acc);
    } else {
        // Wu[k][j], k = (k1,k2,k3) over (4,4,4), j = (j1,j2,j3) over (8,16,8)
        int g = gid - 65536;
        int k = g >> 10, j = g & 1023;
        int k1 = k >> 4, k2 = (k >> 2) & 3,  k3 = k & 3;
        int j1 = j >> 7, j2 = (j >> 3) & 15, j3 = j & 7;
        float acc = 0.f;
        for (int a = 0; a < 8; ++a) {
            float pa = u1[(k1 * 8 + j1) * 8 + a];
            const float* u2p = u2 + ((a * 4 + k2) * 16 + j2) * 8;
            for (int b = 0; b < 8; ++b)
                acc += pa * u2p[b] * u3[(b * 4 + k3) * 8 + j3];
        }
        int t = j >> 4, np = j & 15;
        int s = k >> 5, r = k & 31, h = r >> 4, q = r & 15;
        int lane = np + 16 * h;
        wu_pack[(((t * 2 + s) * 32 + lane) << 4) + q] = f32_to_bf16_rne(acc);
    }
}

// ---------------------------------------------------------------------------
// Kernel 1: fused  y = relu(x @ Wd + b_d) @ Wu + b_u
// One wave32 per 16-row tile of x; 8 waves / block; 128 blocks = 1024 tiles.
// x is streamed non-temporally (read once); packed weights stay resident in
// WGP$/L2 (regular temporal); y is stored non-temporally (written once).
// ---------------------------------------------------------------------------
__global__ void __launch_bounds__(256)
tt_adapter_kernel(const float* __restrict__ x,
                  const float* __restrict__ bias_down,
                  const float* __restrict__ bias_up,
                  const v16bf* __restrict__ wd_pack,
                  const v16bf* __restrict__ wu_pack,
                  float* __restrict__ y) {
    __shared__ __align__(16) __bf16 zbuf[8][16][64];  // per-wave z tile (bf16)

    const int wl   = threadIdx.x >> 5;   // wave in block
    const int lane = threadIdx.x & 31;
    const int m    = lane & 15;          // row (A) / column (B,C,D) index
    const int h    = lane >> 4;          // half-wave select
    const int tile = blockIdx.x * 8 + wl;

    const float* xrow = x + (size_t)(tile * 16 + m) * 1024;

    // ---------------- GEMM1: z16x64 = x16x1024 @ Wd ----------------
    v8f acc[4];
#pragma unroll
    for (int t = 0; t < 4; ++t)
#pragma unroll
        for (int r = 0; r < 8; ++r) acc[t][r] = 0.f;

    for (int s = 0; s < 32; ++s) {
        // A fragment (contiguous order, matching B pack): lane m+16h holds
        // K = 32s + 16h + [0..15]
        const f4* p0 = reinterpret_cast<const f4*>(xrow + s * 32 + 16 * h);
        const f4* p1 = reinterpret_cast<const f4*>(xrow + s * 32 + 16 * h + 8);
        f4 a0 = __builtin_nontemporal_load(p0);
        f4 a1 = __builtin_nontemporal_load(p0 + 1);
        f4 a2 = __builtin_nontemporal_load(p1);
        f4 a3 = __builtin_nontemporal_load(p1 + 1);
        v16f vf;
#pragma unroll
        for (int i = 0; i < 4; ++i) {
            vf[i]      = a0[i];
            vf[4 + i]  = a1[i];
            vf[8 + i]  = a2[i];
            vf[12 + i] = a3[i];
        }
        // Single vector fptrunc -> hardware packed bf16 converts (RNE)
        v16bf abf = __builtin_convertvector(vf, v16bf);

#pragma unroll
        for (int t = 0; t < 4; ++t) {
            v16bf b = wd_pack[(t * 32 + s) * 32 + lane];
            acc[t] = __builtin_amdgcn_wmma_f32_16x16x32_bf16(
                false, abf, false, b, (short)0, acc[t], false, false);
        }
    }

    // Epilogue 1: + bias_down, ReLU, bf16 -> LDS (C layout: VGPR r = row r+8h, col = m)
#pragma unroll
    for (int t = 0; t < 4; ++t) {
        float bd = bias_down[t * 16 + m];
#pragma unroll
        for (int r = 0; r < 8; ++r) {
            float z = acc[t][r] + bd;
            z = z > 0.f ? z : 0.f;
            zbuf[wl][r + 8 * h][t * 16 + m] = (__bf16)z;
        }
    }
    // Same-wave LDS RAW: drain DS counter before re-reading z.
    asm volatile("s_wait_dscnt 0" ::: "memory");

    // Re-read z as A fragments (K = 64 -> 2 k-steps of 32)
    AFrag a2f[2];
#pragma unroll
    for (int s2 = 0; s2 < 2; ++s2) {
        a2f[s2].q[0] = *reinterpret_cast<const uint4*>(&zbuf[wl][m][s2 * 32 + 16 * h]);
        a2f[s2].q[1] = *reinterpret_cast<const uint4*>(&zbuf[wl][m][s2 * 32 + 16 * h + 8]);
    }

    // ---------------- GEMM2: y16x1024 = z16x64 @ Wu + b_u ----------------
    for (int t2 = 0; t2 < 64; ++t2) {
        float bu = bias_up[t2 * 16 + m];
        v8f c;
#pragma unroll
        for (int r = 0; r < 8; ++r) c[r] = bu;   // D = A*B + C gives bias for free
#pragma unroll
        for (int s2 = 0; s2 < 2; ++s2) {
            v16bf b = wu_pack[(t2 * 2 + s2) * 32 + lane];
            c = __builtin_amdgcn_wmma_f32_16x16x32_bf16(
                false, a2f[s2].bf, false, b, (short)0, c, false, false);
        }
        float* yp = y + (size_t)(tile * 16 + 8 * h) * 1024 + t2 * 16 + m;
#pragma unroll
        for (int r = 0; r < 8; ++r)
            __builtin_nontemporal_store(c[r], yp + (size_t)r * 1024);
    }
}

// ---------------------------------------------------------------------------
extern "C" void kernel_launch(void* const* d_in, const int* in_sizes, int n_in,
                              void* d_out, int out_size, void* d_ws, size_t ws_size,
                              hipStream_t stream) {
    const float* x  = (const float*)d_in[0];
    const float* d1 = (const float*)d_in[1];
    const float* d2 = (const float*)d_in[2];
    const float* d3 = (const float*)d_in[3];
    const float* u1 = (const float*)d_in[4];
    const float* u2 = (const float*)d_in[5];
    const float* u3 = (const float*)d_in[6];
    const float* bd = (const float*)d_in[7];
    const float* bu = (const float*)d_in[8];
    float* y = (float*)d_out;

    unsigned short* wd_pack = (unsigned short*)d_ws;       // 128 KB
    unsigned short* wu_pack = wd_pack + 65536;             // 128 KB

    // Weight contraction + bf16 fragment packing (131072 threads total)
    tt_pack_kernel<<<512, 256, 0, stream>>>(d1, d2, d3, u1, u2, u3, wd_pack, wu_pack);

    // Fused adapter: 1024 tiles of 16 rows, 1 wave32 per tile, 8 waves/block
    tt_adapter_kernel<<<128, 256, 0, stream>>>(
        x, bd, bu, (const v16bf*)wd_pack, (const v16bf*)wu_pack, y);
}
